// MultiHead_33011118637667
// MI455X (gfx1250) — matrix operand order repaired and checked by reading, not compiled, not run on hardware
//
#include <hip/hip_runtime.h>
#include <hip/hip_bf16.h>
#include <stdint.h>

// ---------------------------------------------------------------------------
// MultiHead attention forward for MI455X (gfx1250), bf16 WMMA pipeline.
// B=4, S=2048, E=2048, H=16, D=128, NHD=2048.
// ---------------------------------------------------------------------------

typedef __attribute__((ext_vector_type(16))) __bf16 bf16x16;
typedef __attribute__((ext_vector_type(8)))  float  f32x8;

union FragU { bf16x16 v; uint4 q[2]; };
union Pack8 { uint4 q; __bf16 b[8]; };

__device__ __forceinline__ __bf16 f2bf(float f) {
  return (__bf16)f;   // native bf16 convert on gfx1250
}

__device__ __forceinline__ void cvt4(__bf16* dst, float4 v) {
  union { __bf16 b[4]; uint2 u; } t;
  t.b[0] = f2bf(v.x); t.b[1] = f2bf(v.y); t.b[2] = f2bf(v.z); t.b[3] = f2bf(v.w);
  *(uint2*)dst = t.u;
}

// A fragment (16x32 bf16, MxK). p0 points at [row0][k0] in a [rows][pitch] array.
// Lanes 0-15: row=lane,  K 0-7 in v0-3, K 16-23 in v4-7.
// Lanes 16-31: row=lane-16, K 8-15 in v0-3, K 24-31 in v4-7.
__device__ __forceinline__ bf16x16 load_afrag(const __bf16* p0, int pitch) {
  const int lane = threadIdx.x & 31;
  const int m = lane & 15, hi = lane >> 4;
  const __bf16* p = p0 + m * pitch + hi * 8;
  FragU f;
  f.q[0] = *(const uint4*)(p);
  f.q[1] = *(const uint4*)(p + 16);
  return f.v;
}

// B fragment (32x16 bf16, KxN) from an [N][K] layout (column n contiguous in K).
// Lanes 0-15: col=lane, K 0-7 in v0-3, K 16-23 in v4-7; lanes 16-31: col=lane-16, K 8-15, K 24-31.
__device__ __forceinline__ bf16x16 load_bfrag(const __bf16* p0, int pitch) {
  const int lane = threadIdx.x & 31;
  const int n = lane & 15, hi = lane >> 4;
  const __bf16* p = p0 + n * pitch + hi * 8;
  FragU f;
  f.q[0] = *(const uint4*)(p);
  f.q[1] = *(const uint4*)(p + 16);
  return f.v;
}

#define WMMA_BF16(a, b, c) \
  __builtin_amdgcn_wmma_f32_16x16x32_bf16(false, (a), false, (b), (short)0, (c), false, false)

// ---------------------------------------------------------------------------
// Kernel 1: QKV projections.  Y = X @ W^T, X:[8192x2048] f32, W:[2048x2048] f32.
// Output stored bf16 in [b, h, s, d] layout.  gridDim.z in {0,1,2} -> q,k,v.
// Double-buffered LDS staging with f32->bf16 conversion in flight.
// ---------------------------------------------------------------------------
__global__ __launch_bounds__(256) void qkv_gemm_kernel(
    const float* __restrict__ x,
    const float* __restrict__ wq, const float* __restrict__ wk,
    const float* __restrict__ wv,
    __bf16* __restrict__ qo, __bf16* __restrict__ ko, __bf16* __restrict__ vo)
{
  __shared__ __align__(16) __bf16 As[2][128][40];
  __shared__ __align__(16) __bf16 Bs[2][128][40];

  const float* W   = (blockIdx.z == 0) ? wq : (blockIdx.z == 1) ? wk : wv;
  __bf16*      Out = (blockIdx.z == 0) ? qo : (blockIdx.z == 1) ? ko : vo;

  const int m0 = blockIdx.y * 128, n0 = blockIdx.x * 128;
  const int tid = threadIdx.x, wave = tid >> 5, lane = tid & 31;
  const int wm = wave >> 1, wn = wave & 1;   // 4 (M) x 2 (N) waves
  const int r15 = lane & 15, hi = lane >> 4;

  f32x8 acc[2][4] = {};

  const int srow = tid >> 1;           // 128 rows, 2 threads per row
  const int sk   = (tid & 1) * 16;     // 16 K elements each
  const float* aBase = x + (size_t)(m0 + srow) * 2048 + sk;
  const float* bBase = W + (size_t)(n0 + srow) * 2048 + sk;

  // Preload tile kt=0 into buffer 0.
#pragma unroll
  for (int j = 0; j < 4; ++j) {
    cvt4(&As[0][srow][sk + j * 4], *(const float4*)(aBase + j * 4));
    cvt4(&Bs[0][srow][sk + j * 4], *(const float4*)(bBase + j * 4));
  }
  __syncthreads();

  for (int kt = 0; kt < 2048; kt += 32) {
    const int cur = (kt >> 5) & 1;
    const bool has_next = (kt + 32) < 2048;

    // Prefetch next tile global->registers (overlaps with WMMA below).
    float4 apre[4], bpre[4];
    if (has_next) {
#pragma unroll
      for (int j = 0; j < 4; ++j) {
        apre[j] = *(const float4*)(aBase + kt + 32 + j * 4);
        bpre[j] = *(const float4*)(bBase + kt + 32 + j * 4);
      }
      if (kt + 64 < 2048) {
        __builtin_prefetch(aBase + kt + 64, 0, 1);
        __builtin_prefetch(bBase + kt + 64, 0, 1);
      }
    }

    // Compute on current buffer.
    bf16x16 af[2], bfr[4];
#pragma unroll
    for (int i = 0; i < 2; ++i) af[i] = load_afrag(&As[cur][wm * 32 + i * 16][0], 40);
#pragma unroll
    for (int j = 0; j < 4; ++j) bfr[j] = load_bfrag(&Bs[cur][wn * 64 + j * 16][0], 40);
#pragma unroll
    for (int i = 0; i < 2; ++i)
#pragma unroll
      for (int j = 0; j < 4; ++j)
        acc[i][j] = WMMA_BF16(af[i], bfr[j], acc[i][j]);

    // Store next tile to the other buffer.
    if (has_next) {
#pragma unroll
      for (int j = 0; j < 4; ++j) {
        cvt4(&As[cur ^ 1][srow][sk + j * 4], apre[j]);
        cvt4(&Bs[cur ^ 1][srow][sk + j * 4], bpre[j]);
      }
    }
    __syncthreads();
  }

  // Epilogue: scatter to [b, h, s, d] bf16.
#pragma unroll
  for (int i = 0; i < 2; ++i) {
#pragma unroll
    for (int j = 0; j < 4; ++j) {
      const int col = n0 + wn * 64 + j * 16 + r15;
      const int h = col >> 7, d = col & 127;
#pragma unroll
      for (int r = 0; r < 8; ++r) {
        const int mrow = m0 + wm * 32 + i * 16 + r + 8 * hi;
        const int b = mrow >> 11, s = mrow & 2047;
        Out[((size_t)(b * 16 + h) * 2048 + s) * 128 + d] = f2bf(acc[i][j][r]);
      }
    }
  }
}

// ---------------------------------------------------------------------------
// Kernel 2: causal flash attention.  Per block: 128 query rows of one (b,h).
// Each of 8 waves owns 16 rows; key blocks of 32 processed with online softmax.
// ---------------------------------------------------------------------------
__global__ __launch_bounds__(256) void attn_kernel(
    const __bf16* __restrict__ Q, const __bf16* __restrict__ K,
    const __bf16* __restrict__ V, __bf16* __restrict__ Ob)
{
  __shared__ __align__(16) __bf16 Kt[32][136];    // [t][d]
  __shared__ __align__(16) __bf16 Vt[128][40];    // [d][t] (transposed)
  __shared__ __align__(16) __bf16 Pt[8][16][40];  // per-wave P tile [row][t]

  const int bh = blockIdx.y;      // b*16 + h
  const int qb = blockIdx.x;      // 128-row query block
  const int tid = threadIdx.x, wave = tid >> 5, lane = tid & 31;
  const int r15 = lane & 15, hi = lane >> 4;
  const size_t base = (size_t)bh * 2048 * 128;
  const int qrow0 = qb * 128 + wave * 16;

  // Load the wave's Q fragments (4 chunks of K=32 over d).
  bf16x16 qf[4];
  {
    const __bf16* qp = Q + base + (size_t)qrow0 * 128;
#pragma unroll
    for (int c = 0; c < 4; ++c) qf[c] = load_afrag(qp + c * 32, 128);
  }

  f32x8 acc[8] = {};              // out: rows (r + 8*hi), cols d = n*16 + r15
  float mrow[8], lrow[8];
#pragma unroll
  for (int r = 0; r < 8; ++r) { mrow[r] = -1e30f; lrow[r] = 0.0f; }

  const float sc = 0.08838834764831845f * 1.4426950408889634f; // 1/sqrt(128)*log2e

  const int kb_end  = qb * 4 + 3;                        // block-wide (barriers)
  const int kb_wave = (qb * 128 + wave * 16 + 15) >> 5;  // causal limit per wave

  for (int kb = 0; kb <= kb_end; ++kb) {
    __syncthreads();
    {
      // Cooperative staging: thread -> key row t, 16 d-elements.
      const int t = tid >> 3, d0 = (tid & 7) * 16;
      const __bf16* kp = K + base + (size_t)(kb * 32 + t) * 128 + d0;
      *(uint4*)&Kt[t][d0]     = *(const uint4*)(kp);
      *(uint4*)&Kt[t][d0 + 8] = *(const uint4*)(kp + 8);
      const __bf16* vp = V + base + (size_t)(kb * 32 + t) * 128 + d0;
      Pack8 v0, v1;
      v0.q = *(const uint4*)(vp);
      v1.q = *(const uint4*)(vp + 8);
#pragma unroll
      for (int j = 0; j < 8; ++j) {
        Vt[d0 + j][t]     = v0.b[j];
        Vt[d0 + 8 + j][t] = v1.b[j];
      }
    }
    __syncthreads();

    if (kb <= kb_wave) {   // uniform per wave -> EXEC stays all-ones
      // --- scores: S[16x32] = Q[16x128] @ K^T ---
      f32x8 s0 = {}, s1 = {};
#pragma unroll
      for (int c = 0; c < 4; ++c) {
        bf16x16 kb0 = load_bfrag(&Kt[0][c * 32], 136);
        bf16x16 kb1 = load_bfrag(&Kt[16][c * 32], 136);
        s0 = WMMA_BF16(qf[c], kb0, s0);
        s1 = WMMA_BF16(qf[c], kb1, s1);
      }

      // --- mask + online softmax (base-2) ---
      const int t0 = kb * 32 + r15, t1 = t0 + 16;
#pragma unroll
      for (int r = 0; r < 8; ++r) {
        const int srowg = qrow0 + r + 8 * hi;
        float x0 = (t0 <= srowg) ? s0[r] * sc : -1e30f;
        float x1 = (t1 <= srowg) ? s1[r] * sc : -1e30f;
        float mx = fmaxf(x0, x1);
#pragma unroll
        for (int msk = 1; msk <= 8; msk <<= 1) mx = fmaxf(mx, __shfl_xor(mx, msk));
        const float mn   = fmaxf(mrow[r], mx);
        const float corr = __builtin_amdgcn_exp2f(mrow[r] - mn);
        const float p0 = __builtin_amdgcn_exp2f(x0 - mn);
        const float p1 = __builtin_amdgcn_exp2f(x1 - mn);
        float rs = p0 + p1;
#pragma unroll
        for (int msk = 1; msk <= 8; msk <<= 1) rs += __shfl_xor(rs, msk);
        lrow[r] = lrow[r] * corr + rs;
        mrow[r] = mn;
#pragma unroll
        for (int n = 0; n < 8; ++n) acc[n][r] *= corr;
        Pt[wave][r + 8 * hi][r15]      = f2bf(p0);
        Pt[wave][r + 8 * hi][16 + r15] = f2bf(p1);
      }

      // --- output: O += P[16x32] @ V[32x128] ---
      bf16x16 pf = load_afrag(&Pt[wave][0][0], 40);
#pragma unroll
      for (int n = 0; n < 8; ++n) {
        bf16x16 vf = load_bfrag(&Vt[n * 16][0], 40);
        acc[n] = WMMA_BF16(pf, vf, acc[n]);
      }
    }
  }

  // Epilogue: normalize (fast rcp, bf16 output tolerance) and write
  // [b, s, h*128 + d] bf16.
  const int b = bh >> 4, h = bh & 15;
  float invl[8];
#pragma unroll
  for (int r = 0; r < 8; ++r) invl[r] = __builtin_amdgcn_rcpf(lrow[r]);
#pragma unroll
  for (int n = 0; n < 8; ++n) {
    const int d = n * 16 + r15;
#pragma unroll
    for (int r = 0; r < 8; ++r) {
      const int s = qrow0 + r + 8 * hi;
      Ob[(size_t)(b * 2048 + s) * 2048 + h * 128 + d] = f2bf(acc[n][r] * invl[r]);
    }
  }
}

// ---------------------------------------------------------------------------
// Kernel 3: output projection.  out = AttnOut(bf16) @ Wp^T + b_proj, f32 out.
// Double-buffered LDS staging.
// ---------------------------------------------------------------------------
__global__ __launch_bounds__(256) void proj_gemm_kernel(
    const __bf16* __restrict__ A, const float* __restrict__ W,
    const float* __restrict__ bias, float* __restrict__ out)
{
  __shared__ __align__(16) __bf16 As[2][128][40];
  __shared__ __align__(16) __bf16 Bs[2][128][40];

  const int m0 = blockIdx.y * 128, n0 = blockIdx.x * 128;
  const int tid = threadIdx.x, wave = tid >> 5, lane = tid & 31;
  const int wm = wave >> 1, wn = wave & 1;
  const int r15 = lane & 15, hi = lane >> 4;

  f32x8 acc[2][4] = {};

  const int srow = tid >> 1, sk = (tid & 1) * 16;
  const __bf16* aBase = A + (size_t)(m0 + srow) * 2048 + sk;
  const float*  bBase = W + (size_t)(n0 + srow) * 2048 + sk;

  // Preload tile kt=0 into buffer 0.
  *(uint4*)&As[0][srow][sk]     = *(const uint4*)(aBase);
  *(uint4*)&As[0][srow][sk + 8] = *(const uint4*)(aBase + 8);
#pragma unroll
  for (int j = 0; j < 4; ++j)
    cvt4(&Bs[0][srow][sk + j * 4], *(const float4*)(bBase + j * 4));
  __syncthreads();

  for (int kt = 0; kt < 2048; kt += 32) {
    const int cur = (kt >> 5) & 1;
    const bool has_next = (kt + 32) < 2048;

    uint4 a0, a1;
    float4 bpre[4];
    if (has_next) {
      a0 = *(const uint4*)(aBase + kt + 32);
      a1 = *(const uint4*)(aBase + kt + 32 + 8);
#pragma unroll
      for (int j = 0; j < 4; ++j)
        bpre[j] = *(const float4*)(bBase + kt + 32 + j * 4);
      if (kt + 64 < 2048) {
        __builtin_prefetch(aBase + kt + 64, 0, 1);
        __builtin_prefetch(bBase + kt + 64, 0, 1);
      }
    }

    bf16x16 af[2], bfr[4];
#pragma unroll
    for (int i = 0; i < 2; ++i) af[i] = load_afrag(&As[cur][wm * 32 + i * 16][0], 40);
#pragma unroll
    for (int j = 0; j < 4; ++j) bfr[j] = load_bfrag(&Bs[cur][wn * 64 + j * 16][0], 40);
#pragma unroll
    for (int i = 0; i < 2; ++i)
#pragma unroll
      for (int j = 0; j < 4; ++j)
        acc[i][j] = WMMA_BF16(af[i], bfr[j], acc[i][j]);

    if (has_next) {
      *(uint4*)&As[cur ^ 1][srow][sk]     = a0;
      *(uint4*)&As[cur ^ 1][srow][sk + 8] = a1;
#pragma unroll
      for (int j = 0; j < 4; ++j)
        cvt4(&Bs[cur ^ 1][srow][sk + j * 4], bpre[j]);
    }
    __syncthreads();
  }

#pragma unroll
  for (int i = 0; i < 2; ++i) {
#pragma unroll
    for (int j = 0; j < 4; ++j) {
      const int col = n0 + wn * 64 + j * 16 + r15;
      const float bv = bias[col];
#pragma unroll
      for (int r = 0; r < 8; ++r) {
        const int mrow = m0 + wm * 32 + i * 16 + r + 8 * hi;
        out[(size_t)mrow * 2048 + col] = acc[i][j][r] + bv;
      }
    }
  }
}

// ---------------------------------------------------------------------------
// Launch: qkv -> attention -> projection.  Workspace: 4 x 32MB bf16 buffers.
// ---------------------------------------------------------------------------
extern "C" void kernel_launch(void* const* d_in, const int* in_sizes, int n_in,
                              void* d_out, int out_size, void* d_ws, size_t ws_size,
                              hipStream_t stream) {
  const float* x  = (const float*)d_in[0];
  const float* wq = (const float*)d_in[1];
  const float* wk = (const float*)d_in[2];
  const float* wv = (const float*)d_in[3];
  const float* wp = (const float*)d_in[4];
  const float* bp = (const float*)d_in[5];
  float* out = (float*)d_out;

  const size_t elems = (size_t)4 * 16 * 2048 * 128;   // 16,777,216 per buffer
  __bf16* q  = (__bf16*)d_ws;
  __bf16* k  = q + elems;
  __bf16* v  = k + elems;
  __bf16* ao = v + elems;

  dim3 gQKV(16, 64, 3);   // N/128, M/128, {q,k,v}
  qkv_gemm_kernel<<<gQKV, 256, 0, stream>>>(x, wq, wk, wv, q, k, v);

  dim3 gATT(16, 64);      // S/128 query blocks, B*H heads
  attn_kernel<<<gATT, 256, 0, stream>>>(q, k, v, ao);

  dim3 gPRJ(16, 64);      // N/128, M/128
  proj_gemm_kernel<<<gPRJ, 256, 0, stream>>>(ao, wp, bp, out);
}
